// Model_21827023798303
// MI455X (gfx1250) — compile-verified, matrix operand order checked by reading, not run on hardware
//
#include <hip/hip_runtime.h>
#include <hip/hip_bf16.h>
#include <stdint.h>

// ---------------- model dimensions ----------------
#define CHAR_DIM 30
#define FILT     3
#define CHAR_OUT 50
#define WORD_DIM 100
#define HIDDEN   256
#define TARGET   50
#define NTPAD    64           // TARGET padded to 64 for WMMA N
#define START_TAG 48
#define STOP_TAG  49
#define NEGV     (-10000.0f)
#define BATCH    64
#define SEQ      256
#define CLEN     16
#define INP      150          // WORD_DIM + CHAR_OUT
#define INPAD    160          // padded K (5 * 32)
#define GATES    1024         // 4*HIDDEN
#define NPRE     2048         // fwd+bwd gates
#define NPOS     (BATCH*SEQ)  // 16384
#define HCAT     512          // 2*HIDDEN

typedef __attribute__((ext_vector_type(16))) _Float16 v16h;
typedef __attribute__((ext_vector_type(8)))  float    v8f;

union Frag { v16h v; uint32_t u[8]; };
union H2   { _Float16 h[2]; uint32_t u; };

__device__ __forceinline__ float sigm(float x) { return 1.0f / (1.0f + __expf(-x)); }

// K index for fragment dword v of a lane within a 32-wide K tile:
// A/B 16-bit layout: VGPR v holds K-pair starting at (v>>2)*16 + hi*8 + (v&3)*2
__device__ __forceinline__ int kof(int v, int hi) {
    return ((v >> 2) << 4) + (hi << 3) + ((v & 3) << 1);
}

// ---------------- B-matrix packing into per-lane fragment-contiguous layout ----
// dst index = ((ktile*NT + ntile)*32 + lane)*8 + v ; value = half2(w[n][k], w[n][k+1])

// w_ih (both dirs): KT=5 (K=160 padded), NT=128 (N=2048)
__global__ __launch_bounds__(256) void pack_wih(const float* __restrict__ wf,
                                                const float* __restrict__ wb,
                                                uint32_t* __restrict__ dst) {
    int idx = blockIdx.x * 256 + threadIdx.x;
    if (idx >= 5 * 128 * 32 * 8) return;
    int v = idx & 7, lane = (idx >> 3) & 31, ntile = (idx >> 8) & 127, kt = idx >> 15;
    int hi = lane >> 4, lm = lane & 15;
    int k = kt * 32 + kof(v, hi);
    int n = ntile * 16 + lm;
    const float* src = (n < GATES) ? wf : wb;
    int nn = n & (GATES - 1);
    H2 p;
    p.h[0] = (_Float16)((k     < INP) ? src[nn * INP + k]     : 0.0f);
    p.h[1] = (_Float16)((k + 1 < INP) ? src[nn * INP + k + 1] : 0.0f);
    dst[idx] = p.u;
}

// w_hh per dir: KT=8 (K=256), NT=64 (N=1024). dst idx = (((dir*8+kt)*64+ntile)*32+lane)*8+v
__global__ __launch_bounds__(256) void pack_whh(const float* __restrict__ wf,
                                                const float* __restrict__ wb,
                                                uint32_t* __restrict__ dst) {
    int idx = blockIdx.x * 256 + threadIdx.x;
    if (idx >= 2 * 8 * 64 * 32 * 8) return;
    int v = idx & 7, lane = (idx >> 3) & 31, ntile = (idx >> 8) & 63;
    int kt = (idx >> 14) & 7, dir = idx >> 17;
    int hi = lane >> 4, lm = lane & 15;
    int k = kt * 32 + kof(v, hi);
    int n = ntile * 16 + lm;
    const float* src = dir ? wb : wf;
    H2 p;
    p.h[0] = (_Float16)src[n * HIDDEN + k];
    p.h[1] = (_Float16)src[n * HIDDEN + k + 1];
    dst[idx] = p.u;
}

// lin_w: KT=16 (K=512), NT=4 (N=64, cols >= TARGET zero)
__global__ __launch_bounds__(256) void pack_lin(const float* __restrict__ lw,
                                                uint32_t* __restrict__ dst) {
    int idx = blockIdx.x * 256 + threadIdx.x;
    if (idx >= 16 * 4 * 32 * 8) return;
    int v = idx & 7, lane = (idx >> 3) & 31, ntile = (idx >> 8) & 3, kt = idx >> 10;
    int hi = lane >> 4, lm = lane & 15;
    int k = kt * 32 + kof(v, hi);
    int n = ntile * 16 + lm;
    H2 p;
    if (n < TARGET) {
        p.h[0] = (_Float16)lw[n * HCAT + k];
        p.h[1] = (_Float16)lw[n * HCAT + k + 1];
    } else { p.h[0] = (_Float16)0.0f; p.h[1] = (_Float16)0.0f; }
    dst[idx] = p.u;
}

// ------- char CNN + embedding gather -> x16 TIME-MAJOR [st*64+b][INPAD] f16 -------
__global__ __launch_bounds__(64) void charcnn_embed(const int* __restrict__ words,
                                                    const int* __restrict__ chars,
                                                    const float* __restrict__ cemb,
                                                    const float* __restrict__ convw,
                                                    const float* __restrict__ convb,
                                                    const float* __restrict__ wemb,
                                                    _Float16* __restrict__ x16) {
    int pos = blockIdx.x;                 // b*SEQ + s (input order)
    int b = pos >> 8, s = pos & 255;
    size_t row = (size_t)(s * BATCH + b); // time-major output row
    int tid = threadIdx.x;
    __shared__ float ce[CLEN][CHAR_DIM];
    for (int i = tid; i < CLEN * CHAR_DIM; i += 64) {
        int ci = i / CHAR_DIM, d = i % CHAR_DIM;
        ce[ci][d] = cemb[chars[pos * CLEN + ci] * CHAR_DIM + d];
    }
    __syncthreads();
    if (tid < CHAR_OUT) {
        int o = tid;
        float bias = convb[o];
        float mx = -1e30f;
        for (int p = 0; p < CLEN - FILT + 1; ++p) {
            float sum = bias;
#pragma unroll
            for (int f = 0; f < FILT; ++f)
                for (int d = 0; d < CHAR_DIM; ++d)
                    sum += ce[p + f][d] * convw[(o * FILT + f) * CHAR_DIM + d];
            mx = fmaxf(mx, sum);
        }
        x16[row * INPAD + o] = (_Float16)fmaxf(mx, 0.0f);
    }
    int wd = words[pos];
    for (int k = tid; k < WORD_DIM; k += 64)
        x16[row * INPAD + CHAR_OUT + k] = (_Float16)wemb[(size_t)wd * WORD_DIM + k];
    if (tid < INPAD - INP)
        x16[row * INPAD + INP + tid] = (_Float16)0.0f;
}

// ------- pre-GEMM: C-fragment-swizzled pre = x16 @ w_ihT + bias -------
// pre_swz index = (((st*128 + ntile)*4 + mt)*32 + lane)*8 + v
// block 256 threads (8 waves), block tile 128x128, wave tile 64x32 (one st per wave)
__global__ __launch_bounds__(256) void pre_gemm(const _Float16* __restrict__ x16,
                                                const uint32_t* __restrict__ wih,
                                                const float* __restrict__ bf,
                                                const float* __restrict__ bb,
                                                float* __restrict__ pre) {
    const uint32_t* xu = (const uint32_t*)x16;   // ld = 80 dwords
    int lane = threadIdx.x & 31, wv = threadIdx.x >> 5;
    int lm = lane & 15, hi = lane >> 4;
    int rowbase = blockIdx.x * 128 + (wv & 1) * 64;   // multiple of 64 => one st
    int colbase = blockIdx.y * 128 + (wv >> 1) * 32;
    int st = rowbase >> 6;

    v8f acc[4][2];
#pragma unroll
    for (int mt = 0; mt < 4; ++mt)
#pragma unroll
        for (int nt = 0; nt < 2; ++nt) acc[mt][nt] = (v8f){0,0,0,0,0,0,0,0};

    for (int kt = 0; kt < INPAD / 32; ++kt) {
        int kb = kt * 32;
        Frag a[4], b[2];
#pragma unroll
        for (int mt = 0; mt < 4; ++mt) {
            int row = rowbase + mt * 16 + lm;
#pragma unroll
            for (int v = 0; v < 8; ++v)
                a[mt].u[v] = xu[(size_t)row * (INPAD / 2) + ((kb + kof(v, hi)) >> 1)];
        }
#pragma unroll
        for (int nt = 0; nt < 2; ++nt) {
            size_t base = (((size_t)kt * 128 + (colbase >> 4) + nt) * 32 + lane) * 8;
#pragma unroll
            for (int v = 0; v < 8; ++v) b[nt].u[v] = wih[base + v];
        }
#pragma unroll
        for (int mt = 0; mt < 4; ++mt)
#pragma unroll
            for (int nt = 0; nt < 2; ++nt)
                acc[mt][nt] = __builtin_amdgcn_wmma_f32_16x16x32_f16(
                    false, a[mt].v, false, b[nt].v, (short)0, acc[mt][nt], false, false);
    }
#pragma unroll
    for (int nt = 0; nt < 2; ++nt) {
        int col = colbase + nt * 16 + lm;
        float bias = (col < GATES) ? bf[col] : bb[col - GATES];
        int ntile = (colbase >> 4) + nt;
#pragma unroll
        for (int mt = 0; mt < 4; ++mt) {
            size_t base = ((((size_t)st * 128 + ntile) * 4 + mt) * 32 + lane) * 8;
#pragma unroll
            for (int v = 0; v < 8; ++v)
                pre[base + v] = acc[mt][nt][v] + bias;   // contiguous per lane
        }
    }
}

// ------- LSTM scan: one workgroup per direction, 16 waves -------
// wave wv owns hidden cols [16*wv, 16*wv+16); computes all 4 gates for them.
__global__ __launch_bounds__(512) void lstm_scan(const float* __restrict__ pre,
                                                 const uint32_t* __restrict__ whh,
                                                 _Float16* __restrict__ hcat) {
    int dir = blockIdx.x;
    int tid = threadIdx.x;
    int lane = tid & 31, wv = tid >> 5;
    int lm = lane & 15, hi = lane >> 4;
    int j0 = wv * 16;

    __shared__ _Float16 hbuf[BATCH][HIDDEN];   // 32 KB
    for (int i = tid; i < BATCH * HIDDEN; i += 512)
        ((_Float16*)hbuf)[i] = (_Float16)0.0f;

    float c[4][8];
#pragma unroll
    for (int mt = 0; mt < 4; ++mt)
#pragma unroll
        for (int v = 0; v < 8; ++v) c[mt][v] = 0.0f;
    __syncthreads();

    for (int t = 0; t < SEQ; ++t) {
        int st = dir ? (SEQ - 1 - t) : t;
        // ---- accumulator init: contiguous fragment loads from swizzled pre ----
        v8f acc[4][4];   // [gate][mtile]
#pragma unroll
        for (int g = 0; g < 4; ++g) {
            int ntile = dir * 64 + g * 16 + wv;
#pragma unroll
            for (int mt = 0; mt < 4; ++mt) {
                size_t base = ((((size_t)st * 128 + ntile) * 4 + mt) * 32 + lane) * 8;
#pragma unroll
                for (int v = 0; v < 8; ++v) acc[g][mt][v] = pre[base + v];
            }
        }
        // ---- prefetch next step's pre fragments (global_prefetch_b8) ----
        if (t + 1 < SEQ) {
            int stn = dir ? (SEQ - 2 - t) : (t + 1);
#pragma unroll
            for (int g = 0; g < 4; ++g) {
                int ntile = dir * 64 + g * 16 + wv;
#pragma unroll
                for (int mt = 0; mt < 4; ++mt) {
                    size_t base = ((((size_t)stn * 128 + ntile) * 4 + mt) * 32 + lane) * 8;
                    __builtin_prefetch(&pre[base], 0, 1);
                }
            }
        }
        // ---- recurrent GEMM: h @ w_hh^T ----
        for (int kk = 0; kk < HIDDEN / 32; ++kk) {
            int kb = kk * 32;
            Frag a[4];
#pragma unroll
            for (int mt = 0; mt < 4; ++mt) {
                int m = mt * 16 + lm;
#pragma unroll
                for (int v = 0; v < 8; ++v)
                    a[mt].u[v] = *(const uint32_t*)&hbuf[m][kb + kof(v, hi)];
            }
#pragma unroll
            for (int g = 0; g < 4; ++g) {
                size_t base = ((((size_t)dir * 8 + kk) * 64 + g * 16 + wv) * 32 + lane) * 8;
                Frag b;
#pragma unroll
                for (int v = 0; v < 8; ++v) b.u[v] = whh[base + v];
#pragma unroll
                for (int mt = 0; mt < 4; ++mt)
                    acc[g][mt] = __builtin_amdgcn_wmma_f32_16x16x32_f16(
                        false, a[mt].v, false, b.v, (short)0, acc[g][mt], false, false);
            }
        }
        __syncthreads();   // all hbuf reads done before overwrite
        // ---- gates -> (c, h); lane-local; h into LDS ----
#pragma unroll
        for (int mt = 0; mt < 4; ++mt)
#pragma unroll
            for (int v = 0; v < 8; ++v) {
                float gi = acc[0][mt][v], gf = acc[1][mt][v];
                float gg = acc[2][mt][v], go = acc[3][mt][v];
                float cc = sigm(gf) * c[mt][v] + sigm(gi) * tanhf(gg);
                c[mt][v] = cc;
                int m = mt * 16 + v + 8 * hi;
                hbuf[m][j0 + lm] = (_Float16)(sigm(go) * tanhf(cc));
            }
        __syncthreads();   // h complete
        // ---- cooperative b128 copy of this step's h to time-major hcat ----
        {
            const uint4* hb = (const uint4*)&hbuf[0][0];          // 64 rows x 32 uint4
            uint4* hc = (uint4*)hcat;                             // row = 64 uint4
            for (int i = tid; i < BATCH * 32; i += 512) {
                int m = i >> 5, ch = i & 31;
                hc[(size_t)(st * BATCH + m) * (HCAT / 8) + dir * (HIDDEN / 8) + ch] =
                    hb[m * 32 + ch];
            }
        }
        // copy reads complete before next overwrite: next overwrite is after the
        // next iteration's first __syncthreads(), which orders it.
    }
}

// ------- emit GEMM: emit_tm[16384][64] = hcat_tm @ lin_wT + lin_b -------
// block 256 threads (8 waves), each wave 64x64 tile, block covers 512 rows
__global__ __launch_bounds__(256) void emit_gemm(const _Float16* __restrict__ hcat,
                                                 const uint32_t* __restrict__ lin,
                                                 const float* __restrict__ lb,
                                                 float* __restrict__ emit) {
    const uint32_t* hu = (const uint32_t*)hcat;  // ld = 256 dwords
    int lane = threadIdx.x & 31, wv = threadIdx.x >> 5;
    int lm = lane & 15, hi = lane >> 4;
    int rowbase = blockIdx.x * 512 + wv * 64;

    v8f acc[4][4];
#pragma unroll
    for (int mt = 0; mt < 4; ++mt)
#pragma unroll
        for (int nt = 0; nt < 4; ++nt) acc[mt][nt] = (v8f){0,0,0,0,0,0,0,0};

    for (int kk = 0; kk < HCAT / 32; ++kk) {
        int kb = kk * 32;
        Frag a[4], b[4];
#pragma unroll
        for (int mt = 0; mt < 4; ++mt) {
            int row = rowbase + mt * 16 + lm;
#pragma unroll
            for (int v = 0; v < 8; ++v)
                a[mt].u[v] = hu[(size_t)row * (HCAT / 2) + ((kb + kof(v, hi)) >> 1)];
        }
#pragma unroll
        for (int nt = 0; nt < 4; ++nt) {
            size_t base = (((size_t)kk * 4 + nt) * 32 + lane) * 8;
#pragma unroll
            for (int v = 0; v < 8; ++v) b[nt].u[v] = lin[base + v];
        }
#pragma unroll
        for (int mt = 0; mt < 4; ++mt)
#pragma unroll
            for (int nt = 0; nt < 4; ++nt)
                acc[mt][nt] = __builtin_amdgcn_wmma_f32_16x16x32_f16(
                    false, a[mt].v, false, b[nt].v, (short)0, acc[mt][nt], false, false);
    }
#pragma unroll
    for (int mt = 0; mt < 4; ++mt)
#pragma unroll
        for (int nt = 0; nt < 4; ++nt) {
            int col = nt * 16 + lm;
            float bias = (col < TARGET) ? lb[col] : 0.0f;
#pragma unroll
            for (int v = 0; v < 8; ++v) {
                int row = rowbase + mt * 16 + v + 8 * hi;
                emit[(size_t)row * NTPAD + col] = acc[mt][nt][v] + bias;
            }
        }
}

// ------- CRF forward + gold + mean loss (single block); emit is time-major -------
__global__ __launch_bounds__(1024) void crf_loss(const float* __restrict__ emit,
                                                 const float* __restrict__ trans,
                                                 const int* __restrict__ labels,
                                                 float* __restrict__ out) {
    int tid = threadIdx.x;
    __shared__ float alpha[2][BATCH][TARGET];
    __shared__ float tr[TARGET][TARGET];
    __shared__ float red[BATCH];
    for (int i = tid; i < TARGET * TARGET; i += 1024)
        ((float*)tr)[i] = trans[i];
    for (int i = tid; i < BATCH * TARGET; i += 1024) {
        int tt = i % TARGET;
        alpha[0][i / TARGET][tt] = (tt == START_TAG) ? 0.0f : NEGV;
    }
    __syncthreads();
    int cur = 0;
    for (int s = 0; s < SEQ; ++s) {
        for (int idx = tid; idx < BATCH * TARGET; idx += 1024) {
            int b = idx / TARGET, tt = idx % TARGET;
            float e = emit[((size_t)s * BATCH + b) * NTPAD + tt];
            float mx = -1e30f;
            for (int tf = 0; tf < TARGET; ++tf)
                mx = fmaxf(mx, alpha[cur][b][tf] + tr[tt][tf]);
            float sm = 0.0f;
            for (int tf = 0; tf < TARGET; ++tf)
                sm += __expf(alpha[cur][b][tf] + tr[tt][tf] - mx);
            alpha[cur ^ 1][b][tt] = mx + __logf(sm) + e;
        }
        __syncthreads();
        cur ^= 1;
    }
    if (tid < BATCH) {
        int b = tid;
        float mx = -1e30f;
        for (int t = 0; t < TARGET; ++t)
            mx = fmaxf(mx, alpha[cur][b][t] + tr[STOP_TAG][t]);
        float sm = 0.0f;
        for (int t = 0; t < TARGET; ++t)
            sm += __expf(alpha[cur][b][t] + tr[STOP_TAG][t] - mx);
        float logz = mx + __logf(sm);
        int prev = START_TAG;
        float gold = 0.0f;
        for (int s = 0; s < SEQ; ++s) {
            int lab = labels[b * SEQ + s];
            gold += tr[lab][prev];
            gold += emit[((size_t)s * BATCH + b) * NTPAD + lab];
            prev = lab;
        }
        gold += tr[STOP_TAG][prev];
        red[b] = logz - gold;
    }
    __syncthreads();
    if (tid == 0) {
        float s = 0.0f;
        for (int b = 0; b < BATCH; ++b) s += red[b];
        out[0] = s / (float)BATCH;
    }
}

// ---------------- launcher ----------------
extern "C" void kernel_launch(void* const* d_in, const int* in_sizes, int n_in,
                              void* d_out, int out_size, void* d_ws, size_t ws_size,
                              hipStream_t stream) {
    (void)in_sizes; (void)n_in; (void)out_size; (void)ws_size;
    const int*   words  = (const int*)d_in[0];
    const int*   chars  = (const int*)d_in[1];
    const int*   labels = (const int*)d_in[2];
    const float* cemb   = (const float*)d_in[3];
    const float* convw  = (const float*)d_in[4];
    const float* convb  = (const float*)d_in[5];
    const float* wemb   = (const float*)d_in[6];
    const float* w_ih_f = (const float*)d_in[7];
    const float* w_hh_f = (const float*)d_in[8];
    const float* b_f    = (const float*)d_in[9];
    const float* w_ih_b = (const float*)d_in[10];
    const float* w_hh_b = (const float*)d_in[11];
    const float* b_b    = (const float*)d_in[12];
    const float* lin_w  = (const float*)d_in[13];
    const float* lin_b  = (const float*)d_in[14];
    const float* trans  = (const float*)d_in[15];

    char* ws = (char*)d_ws;
    _Float16* x16   = (_Float16*)ws; ws += (size_t)NPOS * INPAD * 2;       // time-major
    uint32_t* wih   = (uint32_t*)ws; ws += (size_t)5 * 128 * 32 * 8 * 4;   // swizzled B
    uint32_t* whh   = (uint32_t*)ws; ws += (size_t)2 * 8 * 64 * 32 * 8 * 4;
    uint32_t* lin   = (uint32_t*)ws; ws += (size_t)16 * 4 * 32 * 8 * 4;
    float*    pre   = (float*)ws;    ws += (size_t)NPOS * NPRE * 4;        // frag-swizzled
    _Float16* hcat  = (_Float16*)ws; ws += (size_t)NPOS * HCAT * 2;       // time-major
    float*    emit  = (float*)ws;    ws += (size_t)NPOS * NTPAD * 4;      // time-major

    pack_wih<<<(5 * 128 * 32 * 8 + 255) / 256, 256, 0, stream>>>(w_ih_f, w_ih_b, wih);
    pack_whh<<<(2 * 8 * 64 * 32 * 8 + 255) / 256, 256, 0, stream>>>(w_hh_f, w_hh_b, whh);
    pack_lin<<<(16 * 4 * 32 * 8 + 255) / 256, 256, 0, stream>>>(lin_w, lin);
    charcnn_embed<<<NPOS, 64, 0, stream>>>(words, chars, cemb, convw, convb, wemb, x16);
    pre_gemm<<<dim3(NPOS / 128, NPRE / 128), 256, 0, stream>>>(x16, wih, b_f, b_b, pre);
    lstm_scan<<<2, 512, 0, stream>>>(pre, whh, hcat);
    emit_gemm<<<NPOS / 512, 256, 0, stream>>>(hcat, lin, lin_b, emit);
    crf_loss<<<1, 1024, 0, stream>>>(emit, trans, labels, (float*)d_out);
}
